// TransformerBasedTransformCodingHyper_77421080478300
// MI455X (gfx1250) — matrix operand order repaired and checked by reading, not compiled, not run on hardware
//
#include <hip/hip_runtime.h>
#include <hip/hip_bf16.h>

// ---------------------------------------------------------------------------
// Swin block on gfx1250 (MI455X): bf16 WMMA everywhere, f32 accumulate.
// ---------------------------------------------------------------------------

typedef __attribute__((ext_vector_type(16))) __bf16 v16bf;
typedef __attribute__((ext_vector_type(8)))  float  v8f;

#define B_    16
#define Hh    56
#define Ww    56
#define Cc    256
#define WS    7
#define SHIFT 3
#define HEADS 8
#define HD    32
#define Nn    49          // tokens per window
#define NWIN  64          // windows per image (8x8)
#define BWIN  (B_*NWIN)   // 1024
#define ROWS  (B_*Hh*Ww)  // 50176 window tokens == image tokens
#define MLPH  1024
#define SCALE 0.17677669529663689f  // 32^-0.5

// A-fragment K index for 16-bit 16x32 A matrix (per ISA 7.12.2):
// lanes 0-15 hold K {0..7,16..23}; lanes 16-31 hold K {8..15,24..31}
__device__ __forceinline__ int kidx(int e, int half) {
  return e + half * 8 + ((e >= 8) ? 8 : 0);
}

__device__ __forceinline__ float wred_max16(float v) {
  v = fmaxf(v, __shfl_xor(v, 1, 32));
  v = fmaxf(v, __shfl_xor(v, 2, 32));
  v = fmaxf(v, __shfl_xor(v, 4, 32));
  v = fmaxf(v, __shfl_xor(v, 8, 32));
  return v;
}
__device__ __forceinline__ float wred_sum16(float v) {
  v += __shfl_xor(v, 1, 32);
  v += __shfl_xor(v, 2, 32);
  v += __shfl_xor(v, 4, 32);
  v += __shfl_xor(v, 8, 32);
  return v;
}
__device__ __forceinline__ float wred_sum32(float v) {
  v += __shfl_xor(v, 16, 32);
  v += __shfl_xor(v, 8, 32);
  v += __shfl_xor(v, 4, 32);
  v += __shfl_xor(v, 2, 32);
  v += __shfl_xor(v, 1, 32);
  return v;
}

// ---------------------------------------------------------------------------
// f32 -> bf16 conversion (weights)
// ---------------------------------------------------------------------------
__global__ __launch_bounds__(256) void f2bf_kernel(const float* __restrict__ in,
                                                   __bf16* __restrict__ out, int n) {
  int i = blockIdx.x * 256 + threadIdx.x;
  if (i < n) out[i] = (__bf16)in[i];
}

// ---------------------------------------------------------------------------
// rowmap LUT: window-token row -> image-token row (window reverse + roll(+3))
// ---------------------------------------------------------------------------
__global__ __launch_bounds__(256) void rowmap_kernel(int* __restrict__ rowmap) {
  int row = blockIdx.x * 256 + threadIdx.x;
  int bw = row / Nn, n = row % Nn;
  int b = bw >> 6, widx = bw & 63;
  int wh = widx >> 3, ww = widx & 7;
  int rr = n / WS, cc = n % WS;
  int hs = wh * WS + rr + SHIFT; if (hs >= Hh) hs -= Hh;
  int wsd = ww * WS + cc + SHIFT; if (wsd >= Ww) wsd -= Ww;
  rowmap[row] = b * (Hh * Ww) + hs * Ww + wsd;
}

// ---------------------------------------------------------------------------
// Combined attention bias (rpb + shift mask, -1e30 padding) precomputed in
// C-fragment layout: pb[nw][head][tile(tm*4+tn)][lane][r]  (f32)
//   i = tm*16 + r + 8*(lane>>4), j = tn*16 + (lane&15)
// ---------------------------------------------------------------------------
__global__ __launch_bounds__(256) void bias_pre_kernel(const float* __restrict__ rpb,
                                                       const float* __restrict__ mask,
                                                       float* __restrict__ pb) {
  int nw = blockIdx.x >> 3;
  int head = blockIdx.x & 7;
  float* dst = pb + (size_t)blockIdx.x * 4096;
  for (int idx = threadIdx.x; idx < 4096; idx += 256) {
    int tile = idx >> 8;
    int tm = tile >> 2, tn = tile & 3;
    int lane = (idx >> 3) & 31;
    int r = idx & 7;
    int i = tm * 16 + r + 8 * (lane >> 4);
    int j = tn * 16 + (lane & 15);
    float v = -1e30f;
    if (i < Nn && j < Nn) {
      int dh = i / WS - j / WS + (WS - 1);
      int dw = i % WS - j % WS + (WS - 1);
      v = rpb[(dh * (2 * WS - 1) + dw) * HEADS + head] +
          mask[(size_t)nw * Nn * Nn + i * Nn + j];
    }
    dst[idx] = v;
  }
}

// ---------------------------------------------------------------------------
// LN1 fused with roll(-3,-3) + window partition -> bf16 in window-token order
// ---------------------------------------------------------------------------
__global__ __launch_bounds__(256) void ln1_shift_kernel(const float* __restrict__ x,
                                                        const float* __restrict__ g,
                                                        const float* __restrict__ bta,
                                                        __bf16* __restrict__ out) {
  int wave = threadIdx.x >> 5;
  int lane = threadIdx.x & 31;
  int row = blockIdx.x * 8 + wave;  // image token index
  const float* xr = x + (size_t)row * Cc;

  float v[8];
  float s = 0.f;
#pragma unroll
  for (int e = 0; e < 8; ++e) { v[e] = xr[e * 32 + lane]; s += v[e]; }
  float mean = wred_sum32(s) * (1.0f / Cc);
  float q = 0.f;
#pragma unroll
  for (int e = 0; e < 8; ++e) { float d = v[e] - mean; q += d * d; }
  float rstd = rsqrtf(wred_sum32(q) * (1.0f / Cc) + 1e-5f);

  int b  = row / (Hh * Ww);
  int hw = row % (Hh * Ww);
  int h  = hw / Ww, w = hw % Ww;
  int hs = h - SHIFT; if (hs < 0) hs += Hh;  // roll(-3)
  int ws = w - SHIFT; if (ws < 0) ws += Ww;
  int wh = hs / WS, r = hs % WS;
  int ww = ws / WS, c = ws % WS;
  int wrow = (b * NWIN + wh * 8 + ww) * Nn + r * WS + c;
  __bf16* orow = out + (size_t)wrow * Cc;
#pragma unroll
  for (int e = 0; e < 8; ++e) {
    int ch = e * 32 + lane;
    orow[ch] = (__bf16)((v[e] - mean) * rstd * g[ch] + bta[ch]);
  }
}

// plain LN (post-attention) -> bf16
__global__ __launch_bounds__(256) void ln2_kernel(const float* __restrict__ x,
                                                  const float* __restrict__ g,
                                                  const float* __restrict__ bta,
                                                  __bf16* __restrict__ out) {
  int wave = threadIdx.x >> 5;
  int lane = threadIdx.x & 31;
  int row = blockIdx.x * 8 + wave;
  const float* xr = x + (size_t)row * Cc;
  float v[8];
  float s = 0.f;
#pragma unroll
  for (int e = 0; e < 8; ++e) { v[e] = xr[e * 32 + lane]; s += v[e]; }
  float mean = wred_sum32(s) * (1.0f / Cc);
  float q = 0.f;
#pragma unroll
  for (int e = 0; e < 8; ++e) { float d = v[e] - mean; q += d * d; }
  float rstd = rsqrtf(wred_sum32(q) * (1.0f / Cc) + 1e-5f);
  __bf16* orow = out + (size_t)row * Cc;
#pragma unroll
  for (int e = 0; e < 8; ++e) {
    int ch = e * 32 + lane;
    orow[ch] = (__bf16)((v[e] - mean) * rstd * g[ch] + bta[ch]);
  }
}

// ---------------------------------------------------------------------------
// bf16 WMMA GEMM: out[M,N] = A[M,K] @ W[N,K]^T + bias, with epilogues.
//   EPI 0: store bf16
//   EPI 1: exact GELU, store bf16
//   EPI 2: rowmap remap + residual(x at remapped row) -> f32
//   EPI 3: + residual(res, same row) -> f32
// Block: 256 thr = 8 waves (2x4), each wave owns 32x32 (2x2 wmma tiles).
// Tile BM=64, BN=128, BK=32.  All staging via 16-byte vector loads.
// ---------------------------------------------------------------------------
#define BM 64
#define BN 128
#define BK 32

template <int EPI>
__global__ __launch_bounds__(256) void gemm_bf16_kernel(
    const __bf16* __restrict__ A, const __bf16* __restrict__ W,
    const float* __restrict__ bias, int K, __bf16* __restrict__ outb, int ldo,
    const float* __restrict__ res, float* __restrict__ outf,
    const int* __restrict__ rowmap) {
  __shared__ alignas(16) __bf16 sA[BM * BK];
  __shared__ alignas(16) __bf16 sB[BN * BK];

  const int tid = threadIdx.x;
  const int wave = tid >> 5;
  const int ll = tid & 31;
  const int half = ll >> 4;
  const int lm = ll & 15;
  const int wm = wave & 1;   // 2 waves along M
  const int wn = wave >> 1;  // 4 waves along N
  const int m0 = blockIdx.x * BM;
  const int n0 = blockIdx.y * BN;

  v8f acc[2][2] = {};

  for (int k0 = 0; k0 < K; k0 += BK) {
    // stage A (64x32) and B (128x32): one/two b128 loads per thread
    {
      int r = tid >> 2, c = (tid & 3) * 8;
      *(uint4*)&sA[r * BK + c] =
          *(const uint4*)&A[(size_t)(m0 + r) * K + k0 + c];
#pragma unroll
      for (int p = 0; p < 2; ++p) {
        int i = (tid + p * 256) * 8;
        int rb = i >> 5, cb = i & 31;
        *(uint4*)&sB[rb * BK + cb] =
            *(const uint4*)&W[(size_t)(n0 + rb) * K + k0 + cb];
      }
    }
    if (k0 + BK < K) {  // prefetch next K tile -> global_prefetch_b8
      __builtin_prefetch(&A[(size_t)(m0 + (tid >> 2)) * K + k0 + BK], 0, 3);
      __builtin_prefetch(&W[(size_t)(n0 + (tid >> 1)) * K + k0 + BK], 0, 3);
    }
    __syncthreads();

    v16bf af[2], bf[2];
#pragma unroll
    for (int mt = 0; mt < 2; ++mt) {
      int row = wm * 32 + mt * 16 + lm;
#pragma unroll
      for (int e = 0; e < 16; ++e) af[mt][e] = sA[row * BK + e + 16 * half];
    }
#pragma unroll
    for (int nt = 0; nt < 2; ++nt) {
      int row = wn * 32 + nt * 16 + lm;
#pragma unroll
      for (int e = 0; e < 16; ++e) bf[nt][e] = sB[row * BK + e + 16 * half];
    }
#pragma unroll
    for (int mt = 0; mt < 2; ++mt)
#pragma unroll
      for (int nt = 0; nt < 2; ++nt)
        acc[mt][nt] = __builtin_amdgcn_wmma_f32_16x16x32_bf16(
            false, af[mt], false, bf[nt], (short)0, acc[mt][nt], false, false);
    __syncthreads();
  }

  // epilogue
  float bv[2];
#pragma unroll
  for (int nt = 0; nt < 2; ++nt) bv[nt] = bias[n0 + wn * 32 + nt * 16 + lm];

#pragma unroll
  for (int mt = 0; mt < 2; ++mt) {
#pragma unroll
    for (int r = 0; r < 8; ++r) {
      int row = m0 + wm * 32 + mt * 16 + r + 8 * half;
      int orow = row;
      if constexpr (EPI == 2) orow = rowmap[row];
#pragma unroll
      for (int nt = 0; nt < 2; ++nt) {
        int col = n0 + wn * 32 + nt * 16 + lm;
        float v = acc[mt][nt][r] + bv[nt];
        if constexpr (EPI == 0) {
          outb[(size_t)row * ldo + col] = (__bf16)v;
        } else if constexpr (EPI == 1) {
          v = 0.5f * v * (1.0f + erff(v * 0.70710678118654752f));
          outb[(size_t)row * ldo + col] = (__bf16)v;
        } else if constexpr (EPI == 2) {
          outf[(size_t)orow * Cc + col] = v + res[(size_t)orow * Cc + col];
        } else {  // EPI == 3
          outf[(size_t)row * Cc + col] = v + res[(size_t)row * Cc + col];
        }
      }
    }
  }
}

// ---------------------------------------------------------------------------
// Windowed attention: one block per window (1024), one wave per head (8).
// N=49 padded to 64; hd=32 -> one bf16 WMMA K-step for Q.K^T.
// Dynamic LDS per wave: Q 64x32, K 64x32, V^T 32x64, P 16x64 (bf16, 14KB).
// V stored transposed so O-stage B-fragments are contiguous (ds_load_b128).
// K and V^T B-fragments are tm-invariant and hoisted out of the tm loop.
// Bias (rpb+mask+padding) pre-baked in fragment layout -> v8f loads.
// ---------------------------------------------------------------------------
__global__ __launch_bounds__(256) void attn_kernel(const __bf16* __restrict__ qkv,
                                                   const float* __restrict__ pb,
                                                   __bf16* __restrict__ attn_out) {
  extern __shared__ __bf16 lds[];
  const int bwin = blockIdx.x;
  const int nw = bwin & (NWIN - 1);
  const int wave = threadIdx.x >> 5;  // head
  const int ll = threadIdx.x & 31;
  const int half = ll >> 4;
  const int lm = ll & 15;
  const int head = wave;

  __bf16* base = lds + wave * 7168;
  __bf16* ldsQ  = base;           // [64][32]
  __bf16* ldsK  = base + 2048;    // [64][32]
  __bf16* ldsVt = base + 4096;    // [32][64]  (transposed V)
  __bf16* ldsP  = base + 6144;    // [16][64]

  const size_t tokbase = (size_t)bwin * Nn;
  const float* pbh = pb + ((size_t)nw * HEADS + head) * 4096 + ll * 8;

  // stage Q,K zero-padded (rows 49..63): 16-byte vector loads, per-wave LDS
  for (int i = ll * 8; i < 2 * 2048; i += 256) {
    int m = i >> 11;  // 0=Q, 1=K
    int rem = i & 2047;
    int j = rem >> 5, d = rem & 31;
    uint4 val{};
    if (j < Nn)
      val = *(const uint4*)&qkv[(tokbase + j) * (3 * Cc) + m * Cc + head * HD + d];
    *(uint4*)&base[i] = val;
  }
  // stage V transposed: Vt[d][j] (zero-padded columns j>=49)
  for (int i = ll * 8; i < 2048; i += 256) {
    int j = i >> 5, d = i & 31;
    uint4 val{};
    if (j < Nn)
      val = *(const uint4*)&qkv[(tokbase + j) * (3 * Cc) + 2 * Cc + head * HD + d];
    __bf16 tmp[8];
    *(uint4*)tmp = val;
#pragma unroll
    for (int t = 0; t < 8; ++t) ldsVt[(d + t) * 64 + j] = tmp[t];
  }

  // hoisted tm-invariant B-fragments
  v16bf bk[4];       // S-stage: B[k=d][n=j] == K[j][d]
#pragma unroll
  for (int tn = 0; tn < 4; ++tn)
#pragma unroll
    for (int e = 0; e < 16; ++e)
      bk[tn][e] = ldsK[(tn * 16 + lm) * HD + e + 16 * half];

  v16bf bvf[2][2];   // O-stage: B[k=j][n=d] == Vt[d][j], contiguous in e
#pragma unroll
  for (int nb = 0; nb < 2; ++nb)
#pragma unroll
    for (int kk = 0; kk < 2; ++kk)
#pragma unroll
      for (int e = 0; e < 16; ++e)
        bvf[nb][kk][e] = ldsVt[(nb * 16 + lm) * 64 + kk * 32 + e + 16 * half];

  const v8f vzero = {};

  for (int tm = 0; tm < 4; ++tm) {
    // Q A-fragment from LDS
    v16bf aq;
#pragma unroll
    for (int e = 0; e < 16; ++e)
      aq[e] = ldsQ[(tm * 16 + lm) * HD + e + 16 * half];

    // S = Q @ K^T : 4 tiles along j
    v8f s[4];
#pragma unroll
    for (int tn = 0; tn < 4; ++tn)
      s[tn] = __builtin_amdgcn_wmma_f32_16x16x32_bf16(false, aq, false, bk[tn],
                                                      (short)0, vzero, false, false);

    // scale + pre-baked (rpb + mask + padding) in fragment layout
#pragma unroll
    for (int tn = 0; tn < 4; ++tn) {
      v8f pbv = *(const v8f*)(pbh + (tm * 4 + tn) * 256);
      s[tn] = s[tn] * SCALE + pbv;
    }

    // row softmax (a row lives in one VGPR index across one 16-lane half)
#pragma unroll
    for (int r = 0; r < 8; ++r) {
      float mx = fmaxf(fmaxf(s[0][r], s[1][r]), fmaxf(s[2][r], s[3][r]));
      mx = wred_max16(mx);
      float sum = 0.f;
#pragma unroll
      for (int tn = 0; tn < 4; ++tn) {
        float e = __expf(s[tn][r] - mx);
        s[tn][r] = e;
        sum += e;
      }
      sum = wred_sum16(sum);
      float inv = 1.0f / sum;
#pragma unroll
      for (int tn = 0; tn < 4; ++tn)
        ldsP[(r + 8 * half) * 64 + tn * 16 + lm] = (__bf16)(s[tn][r] * inv);
    }

    // O = P @ V : 2 output tiles (d blocks), K=64 -> 2 wmma K-steps
#pragma unroll
    for (int nb = 0; nb < 2; ++nb) {
      v8f o = {};
#pragma unroll
      for (int kk = 0; kk < 2; ++kk) {
        v16bf ap;
#pragma unroll
        for (int e = 0; e < 16; ++e)
          ap[e] = ldsP[lm * 64 + kk * 32 + e + 16 * half];
        o = __builtin_amdgcn_wmma_f32_16x16x32_bf16(false, ap, false, bvf[nb][kk],
                                                    (short)0, o, false, false);
      }
#pragma unroll
      for (int r = 0; r < 8; ++r) {
        int i = tm * 16 + r + 8 * half;
        if (i < Nn)
          attn_out[(tokbase + i) * Cc + head * HD + nb * 16 + lm] = (__bf16)o[r];
      }
    }
  }
}

// ---------------------------------------------------------------------------
// Launch
// ---------------------------------------------------------------------------
extern "C" void kernel_launch(void* const* d_in, const int* in_sizes, int n_in,
                              void* d_out, int out_size, void* d_ws, size_t ws_size,
                              hipStream_t stream) {
  const float* x      = (const float*)d_in[0];
  const float* maskm  = (const float*)d_in[1];
  const float* n1g    = (const float*)d_in[2];
  const float* n1b    = (const float*)d_in[3];
  const float* qkv_w  = (const float*)d_in[4];
  const float* qkv_b  = (const float*)d_in[5];
  const float* rpbt   = (const float*)d_in[6];
  const float* proj_w = (const float*)d_in[7];
  const float* proj_b = (const float*)d_in[8];
  const float* n2g    = (const float*)d_in[9];
  const float* n2b    = (const float*)d_in[10];
  const float* fc1_w  = (const float*)d_in[11];
  const float* fc1_b  = (const float*)d_in[12];
  const float* fc2_w  = (const float*)d_in[13];
  const float* fc2_b  = (const float*)d_in[14];
  float* out = (float*)d_out;

  size_t off = 0;
  auto alloc = [&](size_t bytes) -> void* {
    void* p = (char*)d_ws + off;
    off += (bytes + 255) & ~(size_t)255;
    return p;
  };
  __bf16* wqkv  = (__bf16*)alloc((size_t)3 * Cc * Cc * 2);
  __bf16* wproj = (__bf16*)alloc((size_t)Cc * Cc * 2);
  __bf16* wfc1  = (__bf16*)alloc((size_t)MLPH * Cc * 2);
  __bf16* wfc2  = (__bf16*)alloc((size_t)Cc * MLPH * 2);
  int*    rowmap= (int*)alloc((size_t)ROWS * 4);
  float*  pb    = (float*)alloc((size_t)NWIN * HEADS * 4096 * 4);
  __bf16* ln1w  = (__bf16*)alloc((size_t)ROWS * Cc * 2);
  __bf16* qkvb  = (__bf16*)alloc((size_t)ROWS * 3 * Cc * 2);
  __bf16* attno = (__bf16*)alloc((size_t)ROWS * Cc * 2);
  float*  x2    = (float*)alloc((size_t)ROWS * Cc * 4);
  __bf16* ylnb  = (__bf16*)alloc((size_t)ROWS * Cc * 2);
  __bf16* mlph  = (__bf16*)alloc((size_t)ROWS * MLPH * 2);

  // one-time precompute (tiny)
  f2bf_kernel<<<(3 * Cc * Cc + 255) / 256, 256, 0, stream>>>(qkv_w, wqkv, 3 * Cc * Cc);
  f2bf_kernel<<<(Cc * Cc + 255) / 256, 256, 0, stream>>>(proj_w, wproj, Cc * Cc);
  f2bf_kernel<<<(MLPH * Cc + 255) / 256, 256, 0, stream>>>(fc1_w, wfc1, MLPH * Cc);
  f2bf_kernel<<<(Cc * MLPH + 255) / 256, 256, 0, stream>>>(fc2_w, wfc2, Cc * MLPH);
  rowmap_kernel<<<ROWS / 256, 256, 0, stream>>>(rowmap);
  bias_pre_kernel<<<NWIN * HEADS, 256, 0, stream>>>(rpbt, maskm, pb);

  // LN1 + roll + window partition
  ln1_shift_kernel<<<ROWS / 8, 256, 0, stream>>>(x, n1g, n1b, ln1w);

  // QKV: (50176,256) @ (256,768)^T -> bf16
  gemm_bf16_kernel<0><<<dim3(ROWS / BM, (3 * Cc) / BN), 256, 0, stream>>>(
      ln1w, wqkv, qkv_b, Cc, qkvb, 3 * Cc, nullptr, nullptr, nullptr);

  // windowed attention (112 KB dynamic LDS)
  attn_kernel<<<BWIN, 256, 112 * 1024, stream>>>(qkvb, pb, attno);

  // proj + window reverse + roll + residual -> x2 (f32)
  gemm_bf16_kernel<2><<<dim3(ROWS / BM, Cc / BN), 256, 0, stream>>>(
      attno, wproj, proj_b, Cc, nullptr, 0, x, x2, rowmap);

  // LN2
  ln2_kernel<<<ROWS / 8, 256, 0, stream>>>(x2, n2g, n2b, ylnb);

  // FC1 + GELU -> bf16
  gemm_bf16_kernel<1><<<dim3(ROWS / BM, MLPH / BN), 256, 0, stream>>>(
      ylnb, wfc1, fc1_b, Cc, mlph, MLPH, nullptr, nullptr, nullptr);

  // FC2 + residual(x2) -> d_out (f32)
  gemm_bf16_kernel<3><<<dim3(ROWS / BM, Cc / BN), 256, 0, stream>>>(
      mlph, wfc2, fc2_b, MLPH, nullptr, 0, x2, out, nullptr);
}
